// CrossAttention_2302102470735
// MI455X (gfx1250) — matrix operand order repaired and checked by reading, not compiled, not run on hardware
//
#include <hip/hip_runtime.h>
#include <hip/hip_bf16.h>

// ---------------------------------------------------------------------------
// CDNA5 (gfx1250) WMMA implementation of the triple cross-attention block.
//   H=16 heads, DH=64, D=1024, B=8, LQ=4096, LK in {77, 257, 64}
// Heavy math: v_wmma_f32_16x16x32_f16 (wave32).
// Data movement: global_load_async_to_lds_b128 (ASYNCcnt) + double buffering.
// ---------------------------------------------------------------------------

typedef __attribute__((ext_vector_type(16))) _Float16 v16h;
typedef __attribute__((ext_vector_type(8)))  _Float16 v8h;
typedef __attribute__((ext_vector_type(4)))  _Float16 v4h;
typedef __attribute__((ext_vector_type(8)))  float    v8f;

#define DMODEL 1024
#define NHEAD  16
#define DHEAD  64
#define LQTOT  4096
#define BATCH  8

// Build a 16-half fragment from two 16-byte LDS/global chunks.
__device__ __forceinline__ v16h frag2(const _Float16* p0, const _Float16* p1) {
  v8h a = *(const v8h*)p0;
  v8h b = *(const v8h*)p1;
  v16h r;
#pragma unroll
  for (int i = 0; i < 8; ++i) { r[i] = a[i]; r[i + 8] = b[i]; }
  return r;
}

__device__ __forceinline__ v8f wmma16x16x32(v16h a, v16h b, v8f c) {
  return __builtin_amdgcn_wmma_f32_16x16x32_f16(
      /*neg_a=*/false, a, /*neg_b=*/false, b,
      /*c_mod=*/(short)0, c, /*reuse_a=*/false, /*reuse_b=*/false);
}

// Async copy of 16 bytes global -> LDS (tracked with ASYNCcnt).
__device__ __forceinline__ void async_b128(unsigned lds_off, const void* gaddr) {
  asm volatile("global_load_async_to_lds_b128 %0, %1, off"
               :: "v"(lds_off), "v"(gaddr) : "memory");
}
__device__ __forceinline__ void wait_async0() {
  asm volatile("s_wait_asynccnt 0x0" ::: "memory");
}
// Low 32 bits of a generic LDS pointer are the LDS byte offset (aperture is
// 4GB-aligned).
__device__ __forceinline__ unsigned lds_off32(const void* p) {
  return (unsigned)(uintptr_t)p;
}

// ---------------------------------------------------------------------------
// f32 -> f16 conversion (vectorized x4)
// ---------------------------------------------------------------------------
__global__ void __launch_bounds__(256) cvt_f32_f16(const float* __restrict__ in,
                                                   _Float16* __restrict__ out,
                                                   size_t n) {
  size_t i = ((size_t)blockIdx.x * blockDim.x + threadIdx.x) * 4;
  if (i + 4 <= n) {
    float4 v = *(const float4*)(in + i);
    v4h o;
    o[0] = (_Float16)v.x; o[1] = (_Float16)v.y;
    o[2] = (_Float16)v.z; o[3] = (_Float16)v.w;
    *(v4h*)(out + i) = o;
  } else {
    for (size_t j = i; j < n; ++j) out[j] = (_Float16)in[j];
  }
}

// ---------------------------------------------------------------------------
// f32 [1024x1024] -> transposed f16 [1024x1024] (LDS-tiled, 32x32 tiles)
// out[n*1024 + k] = (f16) in[k*1024 + n]
// ---------------------------------------------------------------------------
__global__ void __launch_bounds__(256)
cvt_transpose_f32_f16(const float* __restrict__ in, _Float16* __restrict__ out) {
  __shared__ float tile[32][33];
  const int bx = blockIdx.x * 32;  // source col / dest row
  const int by = blockIdx.y * 32;  // source row / dest col
  const int tx = threadIdx.x & 31;
  const int ty = threadIdx.x >> 5;  // 0..7
#pragma unroll
  for (int i = 0; i < 32; i += 8)
    tile[ty + i][tx] = in[(size_t)(by + ty + i) * DMODEL + bx + tx];
  __syncthreads();
#pragma unroll
  for (int i = 0; i < 32; i += 8)
    out[(size_t)(bx + ty + i) * DMODEL + by + tx] = (_Float16)tile[tx][ty + i];
}

// ---------------------------------------------------------------------------
// WMMA GEMM: C[M,N] = A[M,K] @ W[K,N] (+ optional bias), f16 in, f32 accum.
// W is supplied PRE-TRANSPOSED: Bt[N][K] row-major, so both tiles are plain
// row-major copies staged with global_load_async_to_lds_b128, double-buffered.
// Block: 256 threads = 8 waves; tile BM=128 x BN=128, BK=32.
// Waves 2(m) x 4(n); each wave computes 64x32 via 4x2 WMMA tiles.
// ---------------------------------------------------------------------------
template <typename OutT, bool BIAS>
__global__ void __launch_bounds__(256)
gemm_wmma(const _Float16* __restrict__ A, const _Float16* __restrict__ Bt,
          OutT* __restrict__ C, const float* __restrict__ bias,
          int M, int N, int K) {
  constexpr int BM = 128, BN = 128, BK = 32, PS = 40;  // PS: padded LDS stride
  __shared__ _Float16 As[2][BM * PS];
  __shared__ _Float16 Bs[2][BN * PS];

  const int tid  = threadIdx.x;
  const int wid  = tid >> 5;
  const int lane = tid & 31;
  const int wm = wid >> 2;          // 0..1
  const int wn = wid & 3;           // 0..3
  const int ln = lane & 15;
  const int hb = lane >> 4;         // lane half
  const int m0 = blockIdx.y * BM;
  const int n0 = blockIdx.x * BN;

  const unsigned as0 = lds_off32(&As[0][0]);
  const unsigned bs0 = lds_off32(&Bs[0][0]);

  // Per-thread staging decomposition: 128 rows x 32 halves = 256 thr x 2 x 16B
  const int sr  = tid >> 2;          // 0..63 (and +64)
  const int sc8 = (tid & 3) * 8;     // 0,8,16,24

  auto stage = [&](int buf, int k0) {
#pragma unroll
    for (int it = 0; it < 2; ++it) {
      int rr = sr + it * 64;
      // A tile rows (guarded; untouched LDS rows feed only discarded C rows)
      int gr = m0 + rr;
      if (gr < M)
        async_b128(as0 + 2u * (unsigned)(buf * BM * PS + rr * PS + sc8),
                   A + (size_t)gr * K + k0 + sc8);
      // B tile rows = columns of W (always in range: N,K multiples of 128)
      async_b128(bs0 + 2u * (unsigned)(buf * BN * PS + rr * PS + sc8),
                 Bt + (size_t)(n0 + rr) * K + k0 + sc8);
    }
  };

  v8f acc[4][2];
#pragma unroll
  for (int a = 0; a < 4; ++a)
#pragma unroll
    for (int b = 0; b < 2; ++b) acc[a][b] = (v8f)0.0f;

  stage(0, 0);
  wait_async0();
  __syncthreads();

  const int nk = K / BK;
  for (int ki = 0; ki < nk; ++ki) {
    const int cur = ki & 1;
    if (ki + 1 < nk) stage(cur ^ 1, (ki + 1) * BK);  // prefetch next tile

    v16h bf[2];
#pragma unroll
    for (int nt = 0; nt < 2; ++nt) {
      const _Float16* p = &Bs[cur][(wn * 32 + nt * 16 + ln) * PS + hb * 8];
      bf[nt] = frag2(p, p + 16);
    }
#pragma unroll
    for (int mt = 0; mt < 4; ++mt) {
      const _Float16* p = &As[cur][(wm * 64 + mt * 16 + ln) * PS + hb * 8];
      v16h af = frag2(p, p + 16);
#pragma unroll
      for (int nt = 0; nt < 2; ++nt)
        acc[mt][nt] = wmma16x16x32(af, bf[nt], acc[mt][nt]);
    }
    wait_async0();
    __syncthreads();
  }

  // Epilogue: C layout VGPR r -> row r (lanes 0-15) / r+8 (lanes 16-31)
#pragma unroll
  for (int mt = 0; mt < 4; ++mt)
#pragma unroll
    for (int nt = 0; nt < 2; ++nt)
#pragma unroll
      for (int r = 0; r < 8; ++r) {
        int row = m0 + wm * 64 + mt * 16 + hb * 8 + r;
        if (row < M) {
          int col = n0 + wn * 32 + nt * 16 + ln;
          float v = acc[mt][nt][r];
          if constexpr (BIAS) v += bias[col];
          C[(size_t)row * N + col] = (OutT)v;
        }
      }
}

// ---------------------------------------------------------------------------
// Fused SDPA for one (batch, head, 64-query block), templated on key length.
//   S = (q @ k^T) * 0.125 ; P = softmax(S) ; O = P @ v
//   OutAcc[row, h*64+d]  =  (accumulate ? OutAcc + gain*O : O)
// K rows async-copied to LDS; V transposed into LDS; score buffer reused
// in place as f16 P.
// ---------------------------------------------------------------------------
template <int LK>
__global__ void __launch_bounds__(256)
attn_wmma(const _Float16* __restrict__ Q, const _Float16* __restrict__ Kc,
          const _Float16* __restrict__ Vc, float* __restrict__ OutAcc,
          const float* __restrict__ alpha, int accumulate) {
  constexpr int NT16 = ((LK + 15) / 16) * 16;   // S cols (padded to 16)
  constexpr int NTL  = NT16 / 16;
  constexpr int NT32 = ((LK + 31) / 32) * 32;   // PV K-dim (padded to 32)
  constexpr int KSW  = DHEAD + 8;               // 72 halves, 16B-aligned rows
  constexpr int SW   = NT16 + 4;                // f32 score stride
  constexpr int PW   = SW * 2;                  // f16 P stride (aliases S)
  constexpr int VTW  = NT32 + 8;                // Vt stride
  static_assert(PW >= NT32, "P row capacity");

  __shared__ _Float16 Ks[NT16 * KSW];   // k rows (= B columns for q@k^T)
  __shared__ _Float16 Vt[DHEAD * VTW];  // transposed V: Vt[d][j]
  __shared__ float    S[64 * SW];       // scores; reused as f16 P
  _Float16* P = (_Float16*)S;

  const int tid  = threadIdx.x;
  const int wid  = tid >> 5;
  const int lane = tid & 31;
  const int ln   = lane & 15;
  const int hb   = lane >> 4;
  const int b    = blockIdx.y >> 4;
  const int h    = blockIdx.y & 15;
  const int m0   = blockIdx.x * 64;

  // ---- stage K rows via async copy (pad rows never read downstream) ----
  const unsigned ks0 = lds_off32(&Ks[0]);
  for (int idx = tid; idx < LK * 8; idx += 256) {
    int j = idx >> 3, c8 = (idx & 7) * 8;
    async_b128(ks0 + 2u * (unsigned)(j * KSW + c8),
               Kc + (size_t)(b * LK + j) * DMODEL + h * DHEAD + c8);
  }
  // ---- stage transposed V (zero-pad to NT32: P pad cols are exactly 0) ----
  for (int idx = tid; idx < NT32 * 8; idx += 256) {
    int j = idx >> 3, c8 = (idx & 7) * 8;
    v8h val = (v8h)(_Float16)0.0f;
    if (j < LK)
      val = *(const v8h*)(Vc + (size_t)(b * LK + j) * DMODEL + h * DHEAD + c8);
#pragma unroll
    for (int u = 0; u < 8; ++u) Vt[(c8 + u) * VTW + j] = val[u];
  }
  wait_async0();
  __syncthreads();

  // ---- S = scale * q @ k^T : 4 x NTL tiles over 8 waves ----
  const float scale = 0.125f;  // 64^-0.5
  for (int t = wid; t < 4 * NTL; t += 8) {
    int mt = t / NTL, nt = t % NTL;
    v8f c = (v8f)0.0f;
#pragma unroll
    for (int ks = 0; ks < 2; ++ks) {  // DH=64 => two k=32 steps
      const _Float16* qp = Q + (size_t)(b * LQTOT + m0 + mt * 16 + ln) * DMODEL
                             + h * DHEAD + ks * 32 + hb * 8;
      v16h af = frag2(qp, qp + 16);
      const _Float16* kp = Ks + (nt * 16 + ln) * KSW + ks * 32 + hb * 8;
      v16h bf = frag2(kp, kp + 16);
      c = wmma16x16x32(af, bf, c);
    }
#pragma unroll
    for (int r = 0; r < 8; ++r)
      S[(mt * 16 + hb * 8 + r) * SW + nt * 16 + ln] = c[r] * scale;
  }
  __syncthreads();

  // ---- row softmax; write f16 P in place (forward pass is safe) ----
  if (tid < 64) {
    float* srow = S + tid * SW;
    float mx = -1e30f;
    for (int j = 0; j < LK; ++j) mx = fmaxf(mx, srow[j]);
    float sum = 0.0f;
    for (int j = 0; j < LK; ++j) sum += __expf(srow[j] - mx);
    float inv = 1.0f / sum;
    _Float16* prow = P + tid * PW;
    for (int j = 0; j < NT32; ++j) {
      float e = (j < LK) ? __expf(srow[j] - mx) * inv : 0.0f;
      prow[j] = (_Float16)e;
    }
  }
  __syncthreads();

  // ---- O = P @ V : 4 x 4 tiles over 8 waves (2 each) ----
  float gain = alpha ? (tanhf(alpha[0]) + 1.0f) : 1.0f;
  for (int t = wid; t < 16; t += 8) {
    int mt = t >> 2, nt = t & 3;
    v8f c = (v8f)0.0f;
#pragma unroll
    for (int ks = 0; ks < NT32 / 32; ++ks) {
      const _Float16* pp = P + (mt * 16 + ln) * PW + ks * 32 + hb * 8;
      v16h af = frag2(pp, pp + 16);
      const _Float16* vp = Vt + (nt * 16 + ln) * VTW + ks * 32 + hb * 8;
      v16h bf = frag2(vp, vp + 16);
      c = wmma16x16x32(af, bf, c);
    }
#pragma unroll
    for (int r = 0; r < 8; ++r) {
      size_t row = (size_t)(b * LQTOT + m0 + mt * 16 + hb * 8 + r);
      size_t idx = row * DMODEL + h * DHEAD + nt * 16 + ln;
      float o = c[r] * gain;
      OutAcc[idx] = accumulate ? (OutAcc[idx] + o) : o;
    }
  }
}

// ---------------------------------------------------------------------------
// Host orchestration
// ---------------------------------------------------------------------------
extern "C" void kernel_launch(void* const* d_in, const int* in_sizes, int n_in,
                              void* d_out, int out_size, void* d_ws, size_t ws_size,
                              hipStream_t stream) {
  (void)in_sizes; (void)n_in; (void)out_size; (void)ws_size;
  const float* x       = (const float*)d_in[0];
  const float* ctx_p   = (const float*)d_in[1];
  const float* ctx_i   = (const float*)d_in[2];
  const float* ctx_a   = (const float*)d_in[3];
  const float* Wf[9]   = { (const float*)d_in[4],  (const float*)d_in[5],
                           (const float*)d_in[6],  (const float*)d_in[7],
                           (const float*)d_in[8],  (const float*)d_in[9],
                           (const float*)d_in[10], (const float*)d_in[11],
                           (const float*)d_in[12] };  // Wq Wk Wv Wk_ip Wv_ip Wq_a Wk_a Wv_a Wout
  const float* b_out   = (const float*)d_in[13];
  const float* alpha   = (const float*)d_in[14];
  const float* alpha_a = (const float*)d_in[15];
  float* out = (float*)d_out;

  const int MQ = BATCH * LQTOT;        // 32768
  const int MP = BATCH * 77, MI = BATCH * 257, MA = BATCH * 64;

  char* wsp = (char*)d_ws;
  auto alloc = [&](size_t bytes) -> char* {
    char* p = wsp;
    wsp += (bytes + 255) & ~(size_t)255;
    return p;
  };
  _Float16* Wt[9];  // pre-transposed f16 weights: Wt[n*1024 + k] = W[k,n]
  for (int i = 0; i < 9; ++i) Wt[i] = (_Float16*)alloc((size_t)DMODEL * DMODEL * 2);
  _Float16* Xh  = (_Float16*)alloc((size_t)MQ * DMODEL * 2);
  _Float16* Qh  = (_Float16*)alloc((size_t)MQ * DMODEL * 2);
  _Float16* CPh = (_Float16*)alloc((size_t)MP * DMODEL * 2);
  _Float16* CIh = (_Float16*)alloc((size_t)MI * DMODEL * 2);
  _Float16* CAh = (_Float16*)alloc((size_t)MA * DMODEL * 2);
  _Float16* Kp  = (_Float16*)alloc((size_t)MP * DMODEL * 2);
  _Float16* Vp  = (_Float16*)alloc((size_t)MP * DMODEL * 2);
  _Float16* Kip = (_Float16*)alloc((size_t)MI * DMODEL * 2);
  _Float16* Vip = (_Float16*)alloc((size_t)MI * DMODEL * 2);
  _Float16* Ka  = (_Float16*)alloc((size_t)MA * DMODEL * 2);
  _Float16* Va  = (_Float16*)alloc((size_t)MA * DMODEL * 2);
  float*    ACC = (float*)   alloc((size_t)MQ * DMODEL * 4);

  auto cvt = [&](const float* src, _Float16* dst, size_t n) {
    unsigned g = (unsigned)((n / 4 + 255) / 256);
    cvt_f32_f16<<<dim3(g), dim3(256), 0, stream>>>(src, dst, n);
  };
  auto gemm_h = [&](const _Float16* A, const _Float16* Bt, _Float16* C, int M) {
    dim3 g(DMODEL / 128, (M + 127) / 128);
    gemm_wmma<_Float16, false><<<g, dim3(256), 0, stream>>>(A, Bt, C, nullptr,
                                                            M, DMODEL, DMODEL);
  };

  // 1) conversions: activations straight, weights transposed
  for (int i = 0; i < 9; ++i)
    cvt_transpose_f32_f16<<<dim3(32, 32), dim3(256), 0, stream>>>(Wf[i], Wt[i]);
  cvt(x,     Xh,  (size_t)MQ * DMODEL);
  cvt(ctx_p, CPh, (size_t)MP * DMODEL);
  cvt(ctx_i, CIh, (size_t)MI * DMODEL);
  cvt(ctx_a, CAh, (size_t)MA * DMODEL);

  // 2) projections
  gemm_h(Xh,  Wt[0], Qh,  MQ);   // q = x @ Wq
  gemm_h(CPh, Wt[1], Kp,  MP);   // k
  gemm_h(CPh, Wt[2], Vp,  MP);   // v
  gemm_h(CIh, Wt[3], Kip, MI);   // k_ip
  gemm_h(CIh, Wt[4], Vip, MI);   // v_ip
  gemm_h(CAh, Wt[6], Ka,  MA);   // k_a
  gemm_h(CAh, Wt[7], Va,  MA);   // v_a

  dim3 ag(LQTOT / 64, BATCH * NHEAD);
  // 3) prompt attention -> ACC (overwrite)
  attn_wmma<77><<<ag, dim3(256), 0, stream>>>(Qh, Kp, Vp, ACC, nullptr, 0);
  // 4) image attention, gated by tanh(alpha)+1 -> ACC +=
  attn_wmma<257><<<ag, dim3(256), 0, stream>>>(Qh, Kip, Vip, ACC, alpha, 1);

  // 5) q_a = out @ Wq_a  (reuse Xh/Qh buffers)
  cvt(ACC, Xh, (size_t)MQ * DMODEL);
  gemm_h(Xh, Wt[5], Qh, MQ);

  // 6) action attention, gated -> ACC +=
  attn_wmma<64><<<ag, dim3(256), 0, stream>>>(Qh, Ka, Va, ACC, alpha_a, 1);

  // 7) final projection with bias, f32 output
  cvt(ACC, Xh, (size_t)MQ * DMODEL);
  dim3 g(DMODEL / 128, (MQ + 127) / 128);
  gemm_wmma<float, true><<<g, dim3(256), 0, stream>>>(Xh, Wt[8], out, b_out,
                                                      MQ, DMODEL, DMODEL);
}
